// Sliding_attention_75307956568149
// MI455X (gfx1250) — matrix operand order repaired and checked by reading, not compiled, not run on hardware
//
#include <hip/hip_runtime.h>
#include <cstdint>

#define B_     8
#define CIN    64
#define HW     56
#define NTOK   3136      // 56*56
#define DIN    576       // 64*9
#define COUT   128
#define MT     196       // NTOK/16
#define KT_QKV 18        // DIN/32
#define KT_ATT 98        // NTOK/32
#define NCHUNK 49        // NTOK/64 reduction chunks
#define QSCALE 0.08838834764831845f   // 1/sqrt(128)

typedef __attribute__((ext_vector_type(16))) _Float16 v16h;
typedef __attribute__((ext_vector_type(8)))  _Float16 v8h;
typedef __attribute__((ext_vector_type(8)))  float    v8f;

union F16x16 { v16h v; v8h h[2]; _Float16 e[16]; };

static __device__ __forceinline__ v8f wmma_f16(v16h a, v16h b, v8f c) {
  // D = A(16x32 f16) * B(32x16 f16) + C(16x16 f32)
  return __builtin_amdgcn_wmma_f32_16x16x32_f16(false, a, false, b, (short)0, c, false, false);
}

static __device__ __forceinline__ float gelu_exact(float x) {
  return 0.5f * x * (1.0f + erff(x * 0.70710678118654752440f));
}

// ---------------------------------------------------------------------------
// K1: im2col -> A-fragment-packed f16.  Apack[b][mt][kt][lane][e], 32B/lane.
// A-layout (16x32 f16): M = lane%16, g = lane/16,
//   e in [0,8):  K = kt*32 + 8*g + e ; e in [8,16): K = kt*32 + 16 + 8*g + (e-8)
// Feature order K -> (c, kh, kw)
// ---------------------------------------------------------------------------
__global__ void k_im2col_pack(const float* __restrict__ x, _Float16* __restrict__ Apack) {
  const int TOT = B_ * MT * KT_QKV * 512;
  int idx = blockIdx.x * 256 + threadIdx.x;
  if (idx >= TOT) return;
  int e    = idx & 15;
  int lane = (idx >> 4) & 31;
  int t    = idx >> 9;
  int kt   = t % KT_QKV;
  int mt   = (t / KT_QKV) % MT;
  int b    = t / (KT_QKV * MT);
  int M = lane & 15, g = lane >> 4;
  int K = kt * 32 + ((e < 8) ? (8 * g + e) : (16 + 8 * g + (e - 8)));
  int tok = mt * 16 + M;
  int oh = tok / HW, ow = tok % HW;
  int c = K / 9, r = K % 9, kh = r / 3, kw = r % 3;
  int h = oh + kh - 1, w = ow + kw - 1;
  float val = 0.f;
  if (h >= 0 && h < HW && w >= 0 && w < HW)
    val = x[((b * CIN + c) * HW + h) * HW + w];
  Apack[idx] = (_Float16)val;
}

// ---------------------------------------------------------------------------
// K2: pack Wq|Wk|Wv into B-fragment layout. Wpack[m][kt][nt][lane][e]
// B-layout (32x16 f16): N = lane%16, g = lane/16,
//   e in [0,8):  K = kt*32 + 8*g + e ; e in [8,16): K = kt*32 + 16 + 8*g + (e-8)
// ---------------------------------------------------------------------------
__global__ void k_wpack(const float* __restrict__ Wq, const float* __restrict__ Wk,
                        const float* __restrict__ Wv, _Float16* __restrict__ Wpack) {
  const int TOT = 3 * KT_QKV * 8 * 512;
  int idx = blockIdx.x * 256 + threadIdx.x;
  if (idx >= TOT) return;
  int e    = idx & 15;
  int lane = (idx >> 4) & 31;
  int t    = idx >> 9;
  int nt   = t & 7;
  int kt   = (t >> 3) % KT_QKV;
  int m    = (t >> 3) / KT_QKV;
  int K = kt * 32 + ((e < 8) ? (8 * (lane >> 4) + e) : (16 + 8 * (lane >> 4) + (e - 8)));
  int n = nt * 16 + (lane & 15);
  const float* W = (m == 0) ? Wq : ((m == 1) ? Wk : Wv);
  Wpack[idx] = (_Float16)W[K * COUT + n];
}

// ---------------------------------------------------------------------------
// K3: QKV GEMM (WMMA) + bias + LayerNorm fused. One WG per (b, mt, which).
// q scaled by 1/sqrt(128); q,k stored row-major f16; v stored directly in
// B-fragment-packed layout Vpack[b][kt98][dt][lane][e] for the P@V GEMM.
// ---------------------------------------------------------------------------
__global__ __launch_bounds__(256) void k_qkv(
    const _Float16* __restrict__ Apack, const _Float16* __restrict__ Wpack,
    const float* __restrict__ bq, const float* __restrict__ bk, const float* __restrict__ bv,
    const float* __restrict__ gq, const float* __restrict__ betq,
    const float* __restrict__ gk, const float* __restrict__ betk,
    const float* __restrict__ gv, const float* __restrict__ betv,
    _Float16* __restrict__ qf, _Float16* __restrict__ kf, _Float16* __restrict__ vpack) {
  __shared__ float smem[16 * COUT];
  __shared__ float red[256], red2[256];
  __shared__ float stat[32];

  int wg = blockIdx.x;
  int m  = wg % 3;
  int mt = (wg / 3) % MT;
  int b  = wg / (3 * MT);
  int tid = threadIdx.x, lane = tid & 31, wave = tid >> 5;

  const _Float16* Abase = Apack + ((size_t)(b * MT + mt)) * KT_QKV * 512;
  v8f c = {};
  for (int kt = 0; kt < KT_QKV; ++kt) {
    v16h a  = *(const v16h*)(Abase + kt * 512 + lane * 16);
    v16h bf = *(const v16h*)(Wpack + (((size_t)m * KT_QKV + kt) * 8 + wave) * 512 + lane * 16);
    c = wmma_f16(a, bf, c);
  }
  const float* bias = (m == 0) ? bq : ((m == 1) ? bk : bv);
  int col = wave * 16 + (lane & 15);
  float bcol = bias[col];
  for (int r = 0; r < 8; ++r) {
    int row = r + 8 * (lane >> 4);
    smem[row * COUT + col] = c[r] + bcol;
  }
  __syncthreads();

  // LayerNorm over 128 per row: 16 threads per row
  int row = tid >> 4, sub = tid & 15;
  float s = 0.f, s2 = 0.f;
  for (int i = 0; i < 8; ++i) {
    float xv = smem[row * COUT + sub + 16 * i];
    s += xv; s2 += xv * xv;
  }
  red[tid] = s; red2[tid] = s2;
  __syncthreads();
  if (sub == 0) {
    float S = 0.f, S2 = 0.f;
    for (int i = 0; i < 16; ++i) { S += red[row * 16 + i]; S2 += red2[row * 16 + i]; }
    float mean = S * (1.f / 128.f);
    float var  = S2 * (1.f / 128.f) - mean * mean;
    stat[row * 2] = mean; stat[row * 2 + 1] = rsqrtf(var + 1e-5f);
  }
  __syncthreads();

  const float* gam = (m == 0) ? gq : ((m == 1) ? gk : gv);
  const float* bet = (m == 0) ? betq : ((m == 1) ? betk : betv);
  float scale = (m == 0) ? QSCALE : 1.f;
  for (int i = 0; i < 8; ++i) {
    int eidx = tid + 256 * i;           // 0..2047
    int rr = eidx >> 7, cc = eidx & 127;
    float yv = (smem[eidx] - stat[rr * 2]) * stat[rr * 2 + 1] * gam[cc] + bet[cc];
    if (m == 2) {
      // scatter into B-fragment-packed layout for P@V:
      // value = v[token][cc], token = mt*16+rr
      int token = mt * 16 + rr;
      int kt2   = token >> 5;
      int g2    = (token & 31) >> 4;
      int e2    = token & 15;
      int dt    = cc >> 4;
      int lane2 = g2 * 16 + (cc & 15);
      vpack[((((size_t)b * KT_ATT + kt2) * 8 + dt) * 32 + lane2) * 16 + e2] = (_Float16)yv;
    } else {
      _Float16* out = (m == 0) ? qf : kf;
      out[((size_t)b * NTOK + mt * 16 + rr) * COUT + cc] = (_Float16)(yv * scale);
    }
  }
}

// ---------------------------------------------------------------------------
// K4: fused attention. One WG (8 waves) per (b, 16-row query block).
// LDS: f32 score slab (196KB) + f16 normalized-P slab (98KB) + O (8KB) = 311KB.
// P@V reads P via ds_load_b128 and V via packed 32B v16h global loads.
// ---------------------------------------------------------------------------
__global__ __launch_bounds__(256) void k_attn(
    const _Float16* __restrict__ qf, const _Float16* __restrict__ kf,
    const _Float16* __restrict__ vpack,
    const float* __restrict__ ln1g, const float* __restrict__ ln1b,
    float* __restrict__ ctx) {
  extern __shared__ char smraw[];
  float*    Sbuf   = (float*)smraw;                                    // 16*3136 f32
  _Float16* Pbuf   = (_Float16*)(smraw + 16 * NTOK * 4);               // 16*3136 f16
  float*    Obuf   = (float*)(smraw + 16 * NTOK * 6);                  // 16*128 f32
  float*    red    = Obuf + 16 * COUT;                                 // 256
  float*    red2   = red + 256;                                        // 256
  float*    rowmax = red2 + 256;                                       // 16
  float*    rowinv = rowmax + 16;                                      // 16
  float*    stat   = rowinv + 16;                                      // 32

  int b  = blockIdx.x / MT;
  int mt = blockIdx.x % MT;
  int tid = threadIdx.x, lane = tid & 31, wave = tid >> 5;
  int mlane = lane & 15, g = lane >> 4;

  // q fragments (A role), 4 fragments cover d = 0..127; 1/sqrt(d) pre-folded.
  const _Float16* qrow = qf + ((size_t)b * NTOK + mt * 16 + mlane) * COUT;
  F16x16 qa[4];
  for (int kt = 0; kt < 4; ++kt) {
    qa[kt].h[0] = *(const v8h*)(qrow + kt * 32 + 16 * g);
    qa[kt].h[1] = *(const v8h*)(qrow + kt * 32 + 16 * g + 8);
  }

  // S = q @ k^T : wave w handles key tiles w, w+8, ...
  const _Float16* kb = kf + (size_t)b * NTOK * COUT;
  for (int t = wave; t < MT; t += 8) {
    v8f c = {};
    for (int kt = 0; kt < 4; ++kt) {
      v16h bf = *(const v16h*)(kb + (size_t)(t * 16 + mlane) * COUT + kt * 32 + 16 * g);
      c = wmma_f16(qa[kt].v, bf, c);
    }
    for (int r = 0; r < 8; ++r)
      Sbuf[(r + 8 * g) * NTOK + t * 16 + mlane] = c[r];
  }
  __syncthreads();

  // exact softmax over 3136, 16 threads per row
  int row = tid >> 4, sub = tid & 15;
  float mx = -3.0e38f;
  for (int j = sub; j < NTOK; j += 16) mx = fmaxf(mx, Sbuf[row * NTOK + j]);
  red[tid] = mx;
  __syncthreads();
  if (sub == 0) {
    float m2 = red[row * 16];
    for (int i = 1; i < 16; ++i) m2 = fmaxf(m2, red[row * 16 + i]);
    rowmax[row] = m2;
  }
  __syncthreads();
  float rm = rowmax[row];
  float sum = 0.f;
  for (int j = sub; j < NTOK; j += 16) {
    float ev = __expf(Sbuf[row * NTOK + j] - rm);
    Sbuf[row * NTOK + j] = ev;
    sum += ev;
  }
  red[tid] = sum;
  __syncthreads();
  if (sub == 0) {
    float S = 0.f;
    for (int i = 0; i < 16; ++i) S += red[row * 16 + i];
    rowinv[row] = 1.f / S;
  }
  __syncthreads();

  // normalize P into f16 slab (A-fragment friendly row-major layout)
  float rinv = rowinv[row];
  for (int j = sub; j < NTOK; j += 16)
    Pbuf[row * NTOK + j] = (_Float16)(Sbuf[row * NTOK + j] * rinv);
  __syncthreads();

  // O = P @ V : wave w owns output columns [16w, 16w+16)
  const _Float16* prow = Pbuf + (size_t)mlane * NTOK;
  const _Float16* vbase = vpack + (size_t)b * KT_ATT * 8 * 512 + (size_t)wave * 512 + lane * 16;
  v8f o = {};
  for (int kt = 0; kt < KT_ATT; ++kt) {
    if (kt + 4 < KT_ATT)
      __builtin_prefetch(vbase + (size_t)(kt + 4) * 8 * 512, 0, 0);
    int kbase = kt * 32;
    F16x16 pa;
    pa.h[0] = *(const v8h*)(prow + kbase + 16 * g);         // ds_load_b128
    pa.h[1] = *(const v8h*)(prow + kbase + 16 * g + 8);     // ds_load_b128
    v16h vb = *(const v16h*)(vbase + (size_t)kt * 8 * 512); // 32B packed load
    o = wmma_f16(pa.v, vb, o);
  }
  for (int r = 0; r < 8; ++r)
    Obuf[(r + 8 * g) * COUT + wave * 16 + mlane] = o[r];
  __syncthreads();

  // ln1 + exact GELU epilogue
  float sA = 0.f, sB = 0.f;
  for (int i = 0; i < 8; ++i) {
    float xv = Obuf[row * COUT + sub + 16 * i];
    sA += xv; sB += xv * xv;
  }
  red[tid] = sA; red2[tid] = sB;
  __syncthreads();
  if (sub == 0) {
    float S = 0.f, S2 = 0.f;
    for (int i = 0; i < 16; ++i) { S += red[row * 16 + i]; S2 += red2[row * 16 + i]; }
    float mean = S * (1.f / 128.f);
    float var  = S2 * (1.f / 128.f) - mean * mean;
    stat[row * 2] = mean; stat[row * 2 + 1] = rsqrtf(var + 1e-5f);
  }
  __syncthreads();
  for (int i = 0; i < 8; ++i) {
    int eidx = tid + 256 * i;
    int rr = eidx >> 7, cc = eidx & 127;
    float xv = (Obuf[eidx] - stat[rr * 2]) * stat[rr * 2 + 1] * ln1g[cc] + ln1b[cc];
    ctx[((size_t)b * NTOK + mt * 16 + rr) * COUT + cc] = gelu_exact(xv);
  }
}

// ---------------------------------------------------------------------------
// K5a: coalesced partial reductions for gate / BN.  One WG per (b, 64-token
// chunk); threads map to consecutive channels -> fully coalesced 128-f32 rows.
// ---------------------------------------------------------------------------
__global__ __launch_bounds__(256) void k_gate_partial(
    const float* __restrict__ ctx, const float* __restrict__ Wca,
    float* __restrict__ Gp0, float* __restrict__ Gp1, float* __restrict__ Gp2) {
  __shared__ float r0[256], r1[256], r2[256];
  int chunk = blockIdx.x % NCHUNK;
  int b     = blockIdx.x / NCHUNK;
  int tid = threadIdx.x;
  int d    = tid & 127;
  int half = tid >> 7;                   // 0 or 1
  const float* base = ctx + (size_t)b * NTOK * COUT;
  float a0 = 0.f, a1 = 0.f, a2 = 0.f;
  for (int i = 0; i < 32; ++i) {
    int n = chunk * 64 + half + 2 * i;
    float v = base[(size_t)n * COUT + d];
    a0 += v * Wca[n]; a1 += v; a2 += v * v;
  }
  r0[tid] = a0; r1[tid] = a1; r2[tid] = a2;
  __syncthreads();
  if (half == 0) {
    size_t o = ((size_t)b * NCHUNK + chunk) * COUT + d;
    Gp0[o] = a0 + r0[tid + 128];
    Gp1[o] = a1 + r1[tid + 128];
    Gp2[o] = a2 + r2[tid + 128];
  }
}

// K5b: finalize per-(b,d) gate dot + sums
__global__ void k_gate_final(const float* __restrict__ Gp0, const float* __restrict__ Gp1,
                             const float* __restrict__ Gp2, const float* __restrict__ bca,
                             float* __restrict__ gate, float* __restrict__ S1,
                             float* __restrict__ S2) {
  int idx = blockIdx.x * 256 + threadIdx.x;      // B_*COUT = 1024
  if (idx >= B_ * COUT) return;
  int b = idx >> 7, d = idx & 127;
  float a0 = 0.f, a1 = 0.f, a2 = 0.f;
  for (int c = 0; c < NCHUNK; ++c) {
    size_t o = ((size_t)b * NCHUNK + c) * COUT + d;
    a0 += Gp0[o]; a1 += Gp1[o]; a2 += Gp2[o];
  }
  gate[idx] = a0 + bca[0];
  S1[idx] = a1;
  S2[idx] = a2;
}

// K6: BatchNorm channel stats (training mode, biased var)
__global__ void k_bnstat(const float* __restrict__ gate, const float* __restrict__ S1,
                         const float* __restrict__ S2,
                         float* __restrict__ mu, float* __restrict__ rsig) {
  int d = threadIdx.x;
  float m = 0.f, q = 0.f;
  for (int b = 0; b < B_; ++b) {
    float gv = gate[b * COUT + d];
    m += gv * S1[b * COUT + d];
    q += gv * gv * S2[b * COUT + d];
  }
  const float inv = 1.f / (float)(B_ * NTOK);
  float mean = m * inv;
  float var  = q * inv - mean * mean;
  mu[d] = mean;
  rsig[d] = rsqrtf(var + 1e-5f);
}

// K7: final elementwise gate * BN * GELU, output [B, 128, 56, 56]
__global__ void k_final(const float* __restrict__ ctx, const float* __restrict__ gate,
                        const float* __restrict__ mu, const float* __restrict__ rsig,
                        const float* __restrict__ bng, const float* __restrict__ bnb,
                        float* __restrict__ out) {
  const int TOT = B_ * COUT * NTOK;
  int idx = blockIdx.x * 256 + threadIdx.x;
  if (idx >= TOT) return;
  int n = idx % NTOK;
  int d = (idx / NTOK) & 127;
  int b = idx / (NTOK * COUT);
  float y = gate[b * COUT + d] * ctx[((size_t)b * NTOK + n) * COUT + d];
  y = (y - mu[d]) * rsig[d] * bng[d] + bnb[d];
  out[idx] = gelu_exact(y);
}

// ---------------------------------------------------------------------------
extern "C" void kernel_launch(void* const* d_in, const int* in_sizes, int n_in,
                              void* d_out, int out_size, void* d_ws, size_t ws_size,
                              hipStream_t stream) {
  const float* x     = (const float*)d_in[0];
  const float* Wq    = (const float*)d_in[1];
  const float* bq    = (const float*)d_in[2];
  const float* lnqg  = (const float*)d_in[3];
  const float* lnqb  = (const float*)d_in[4];
  const float* Wk    = (const float*)d_in[5];
  const float* bk    = (const float*)d_in[6];
  const float* lnkg  = (const float*)d_in[7];
  const float* lnkb  = (const float*)d_in[8];
  const float* Wv    = (const float*)d_in[9];
  const float* bv    = (const float*)d_in[10];
  const float* lnvg  = (const float*)d_in[11];
  const float* lnvb  = (const float*)d_in[12];
  const float* ln1g  = (const float*)d_in[13];
  const float* ln1b  = (const float*)d_in[14];
  const float* Wca   = (const float*)d_in[15];
  const float* bca   = (const float*)d_in[16];
  const float* bng   = (const float*)d_in[17];
  const float* bnb   = (const float*)d_in[18];
  float* out = (float*)d_out;

  // Workspace carve (256B aligned)
  char* p = (char*)d_ws;
  auto carve = [&](size_t bytes) -> char* {
    char* r = p;
    p += (bytes + 255) & ~(size_t)255;
    return r;
  };
  _Float16* Apack = (_Float16*)carve((size_t)B_ * MT * KT_QKV * 512 * 2);   // 28.9 MB
  _Float16* Wpack = (_Float16*)carve((size_t)3 * KT_QKV * 8 * 512 * 2);     // 0.44 MB
  _Float16* qf    = (_Float16*)carve((size_t)B_ * NTOK * COUT * 2);
  _Float16* kf    = (_Float16*)carve((size_t)B_ * NTOK * COUT * 2);
  _Float16* vpack = (_Float16*)carve((size_t)B_ * KT_ATT * 8 * 512 * 2);    // 6.4 MB
  float*    ctx   = (float*)carve((size_t)B_ * NTOK * COUT * 4);            // 12.8 MB
  float*    Gp0   = (float*)carve((size_t)B_ * NCHUNK * COUT * 4);
  float*    Gp1   = (float*)carve((size_t)B_ * NCHUNK * COUT * 4);
  float*    Gp2   = (float*)carve((size_t)B_ * NCHUNK * COUT * 4);
  float*    gate  = (float*)carve((size_t)B_ * COUT * 4);
  float*    S1    = (float*)carve((size_t)B_ * COUT * 4);
  float*    S2    = (float*)carve((size_t)B_ * COUT * 4);
  float*    mu    = (float*)carve((size_t)COUT * 4);
  float*    rsig  = (float*)carve((size_t)COUT * 4);

  const int TOT_A = B_ * MT * KT_QKV * 512;
  k_im2col_pack<<<(TOT_A + 255) / 256, 256, 0, stream>>>(x, Apack);

  const int TOT_W = 3 * KT_QKV * 8 * 512;
  k_wpack<<<(TOT_W + 255) / 256, 256, 0, stream>>>(Wq, Wk, Wv, Wpack);

  k_qkv<<<B_ * MT * 3, 256, 0, stream>>>(Apack, Wpack, bq, bk, bv,
                                         lnqg, lnqb, lnkg, lnkb, lnvg, lnvb,
                                         qf, kf, vpack);

  // 16*3136*4 (S f32) + 16*3136*2 (P f16) + 16*128*4 (O) + 2*256*4 + 16+16+32 f32
  size_t shmem = (size_t)16 * NTOK * 6 + (size_t)(16 * COUT + 256 + 256 + 16 + 16 + 32) * 4;
  k_attn<<<B_ * MT, 256, shmem, stream>>>(qf, kf, vpack, ln1g, ln1b, ctx);

  k_gate_partial<<<B_ * NCHUNK, 256, 0, stream>>>(ctx, Wca, Gp0, Gp1, Gp2);
  k_gate_final<<<(B_ * COUT + 255) / 256, 256, 0, stream>>>(Gp0, Gp1, Gp2, bca, gate, S1, S2);
  k_bnstat<<<1, COUT, 0, stream>>>(gate, S1, S2, mu, rsig);

  const int TOT_O = B_ * COUT * NTOK;
  k_final<<<(TOT_O + 255) / 256, 256, 0, stream>>>(ctx, gate, mu, rsig, bng, bnb, out);
}
